// VarAnt_46351287059097
// MI455X (gfx1250) — matrix operand order repaired and checked by reading, not compiled, not run on hardware
//
#include <hip/hip_runtime.h>
#include <hip/hip_bf16.h>
#include <cstddef>

// ---------------------------------------------------------------------------
// CDNA5 (gfx1250) implementation of the VRNN-style reference.
// All GEMMs run through V_WMMA_F32_16X16X32_BF16 (f32 data converted to bf16
// while staging tiles into LDS, f32 accumulation).
// ---------------------------------------------------------------------------

typedef __bf16 bf16;
typedef __attribute__((ext_vector_type(16))) __bf16 v16bf;
typedef __attribute__((ext_vector_type(8)))  float  v8f;

#define BM 128
#define BN 64
#define BK 32
#define BKP 40   // padded LDS row in bf16 elems: 80B (16B aligned, bank-spread)

// ---------------- device math helpers ----------------
__device__ __forceinline__ float sp_f(float x) {          // softplus
  return fmaxf(x, 0.f) + log1pf(expf(-fabsf(x)));
}
__device__ __forceinline__ float sig_f(float x) { return 1.f / (1.f + expf(-x)); }
__device__ __forceinline__ float kld_e_f(float m1, float s1, float m2, float s2) {
  float d = m1 - m2;
  return 2.f * logf(s2) - 2.f * logf(s1) + (s1 * s1 + d * d) / (s2 * s2) - 1.f;
}

// ---------------- WMMA GEMM: Y = act(X[M,K] @ W[K,N] + bias) ----------------
// Requires: M % 128 == 0, K % 32 == 0. N arbitrary (guarded).
template<int ACT>  // 0 = none, 1 = relu
__global__ __launch_bounds__(256) void va_gemm(
    const float* __restrict__ X, const float* __restrict__ W,
    const float* __restrict__ bias, float* __restrict__ Y,
    int M, int N, int K)
{
  __shared__ __align__(16) bf16 As[BM][BKP];   // [m][k]
  __shared__ __align__(16) bf16 Bs[BN][BKP];   // transposed: [n][k]

  const int t    = threadIdx.x;
  const int lane = t & 31;
  const int w    = t >> 5;          // 8 waves
  const int wm   = w & 3;           // 4 waves along M
  const int wn   = w >> 2;          // 2 waves along N
  const int half = lane >> 4;       // lane group 0/1
  const int nl   = lane & 15;

  const int m0 = blockIdx.y * BM;
  const int n0 = blockIdx.x * BN;

  v8f acc[2][2] = {};

  for (int k0 = 0; k0 < K; k0 += BK) {
    // ---- stage A tile (128x32 f32 -> bf16), 16 elems / thread ----
    {
      const int tk = (t & 7) << 2;   // k offset 0..28
      const int tm = t >> 3;         // 0..31
#pragma unroll
      for (int i = 0; i < 4; ++i) {
        const int m = tm + (i << 5);
        const float4 v = *(const float4*)(X + (size_t)(m0 + m) * K + k0 + tk);
        As[m][tk]     = (bf16)v.x;
        As[m][tk + 1] = (bf16)v.y;
        As[m][tk + 2] = (bf16)v.z;
        As[m][tk + 3] = (bf16)v.w;
      }
    }
    // ---- stage B tile (32x64 f32 -> bf16, transposed), 8 elems / thread ----
    {
      const int tn = (t & 15) << 2;  // n offset 0..60
      const int tk = t >> 4;         // 0..15
#pragma unroll
      for (int i = 0; i < 2; ++i) {
        const int k = tk + (i << 4);
        const float* src = W + (size_t)(k0 + k) * N + n0 + tn;
        float4 v;
        if (n0 + tn + 3 < N) {
          v = *(const float4*)src;
        } else {
          v.x = (n0 + tn     < N) ? src[0] : 0.f;
          v.y = (n0 + tn + 1 < N) ? src[1] : 0.f;
          v.z = (n0 + tn + 2 < N) ? src[2] : 0.f;
          v.w = (n0 + tn + 3 < N) ? src[3] : 0.f;
        }
        Bs[tn][k]     = (bf16)v.x;
        Bs[tn + 1][k] = (bf16)v.y;
        Bs[tn + 2][k] = (bf16)v.z;
        Bs[tn + 3][k] = (bf16)v.w;
      }
    }
    __syncthreads();

    // ---- build fragments per ISA 7.12.2 layouts ----
    union Frag { float4 f4[2]; v16bf v; };
    Frag a[2], b[2];
#pragma unroll
    for (int mi = 0; mi < 2; ++mi) {
      // A: lane row m = nl; elems 0..7 -> K=half*8.. ; elems 8..15 -> K=16+half*8..
      const float4* row = (const float4*)(&As[wm * 32 + mi * 16 + nl][0]);
      a[mi].f4[0] = row[half];
      a[mi].f4[1] = row[half + 2];
    }
#pragma unroll
    for (int ni = 0; ni < 2; ++ni) {
      // B: lane col n = nl; elems 0..7 -> K=half*8.. ; elems 8..15 -> K=16+half*8..
      const float4* row = (const float4*)(&Bs[wn * 32 + ni * 16 + nl][0]);
      b[ni].f4[0] = row[half];
      b[ni].f4[1] = row[half + 2];
    }
#pragma unroll
    for (int mi = 0; mi < 2; ++mi)
#pragma unroll
      for (int ni = 0; ni < 2; ++ni)
        acc[mi][ni] = __builtin_amdgcn_wmma_f32_16x16x32_bf16(
            false, a[mi].v, false, b[ni].v, (short)0, acc[mi][ni], false, false);
    __syncthreads();
  }

  // ---- store: C/D layout lane nl -> col, VGPR v -> row v + half*8 ----
#pragma unroll
  for (int mi = 0; mi < 2; ++mi) {
#pragma unroll
    for (int ni = 0; ni < 2; ++ni) {
      const int col = n0 + wn * 32 + ni * 16 + nl;
      if (col < N) {
        const float bv = bias[col];
#pragma unroll
        for (int v = 0; v < 8; ++v) {
          const int row = m0 + wm * 32 + mi * 16 + half * 8 + v;
          float y = acc[mi][ni][v] + bv;
          if (ACT == 1) y = fmaxf(y, 0.f);
          Y[(size_t)row * N + col] = y;
        }
      }
    }
  }
}

// ---------------- elementwise / reduction kernels ----------------

// Y[r, 0:K1] = X1[r]; Y[r, K1:K1+K2] = X2[r]
__global__ void va_concat2(const float* __restrict__ X1, int K1,
                           const float* __restrict__ X2, int K2,
                           float* __restrict__ Y, long long total) {
  long long idx = (long long)blockIdx.x * blockDim.x + threadIdx.x;
  if (idx >= total) return;
  const int K = K1 + K2;
  long long r = idx / K;
  int c = (int)(idx - r * K);
  Y[idx] = (c < K1) ? X1[r * K1 + c] : X2[r * K2 + (c - K1)];
}

// Y[o, rep, inner] = X[o, inner]  (broadcast along middle axis)
__global__ void va_bcast_mid(const float* __restrict__ X, float* __restrict__ Y,
                             int rep, long long inner, long long total) {
  long long idx = (long long)blockIdx.x * blockDim.x + threadIdx.x;
  if (idx >= total) return;
  long long o = idx / (rep * inner);
  long long i = idx % inner;
  Y[idx] = X[o * inner + i];
}

// Y[o, rep, inner] = mean[o, inner] + eps[o, rep, inner] * softplus(mean[o, inner])
__global__ void va_reparam(const float* __restrict__ mean, const float* __restrict__ eps,
                           float* __restrict__ Y, int rep, long long inner, long long total) {
  long long idx = (long long)blockIdx.x * blockDim.x + threadIdx.x;
  if (idx >= total) return;
  long long o = idx / (rep * inner);
  long long i = idx % inner;
  float m = mean[o * inner + i];
  Y[idx] = m + eps[idx] * sp_f(m);
}

// GRU gate fusion: h = (1-u)*n + u*h  (gates ordered r, z, n; in-place h)
__global__ void va_gru(const float* __restrict__ gi, const float* __restrict__ gh,
                       float* __restrict__ h, int Bd, int Hd) {
  int idx = blockIdx.x * blockDim.x + threadIdx.x;
  int total = Bd * Hd;
  if (idx >= total) return;
  int b = idx / Hd, c = idx - b * Hd;
  const float* gib = gi + (size_t)b * 3 * Hd;
  const float* ghb = gh + (size_t)b * 3 * Hd;
  float r = sig_f(gib[c] + ghb[c]);
  float u = sig_f(gib[Hd + c] + ghb[Hd + c]);
  float n = tanhf(gib[2 * Hd + c] + r * ghb[2 * Hd + c]);
  h[idx] = (1.f - u) * n + u * h[idx];
}

// out += 0.5 * sum_i kld_e(m1, sp(m1), m2, sp(m2))   (atomic accumulate)
__global__ void va_kld_sum(const float* __restrict__ m1, const float* __restrict__ m2,
                           int n, float* __restrict__ out) {
  float s = 0.f;
  for (int i = blockIdx.x * blockDim.x + threadIdx.x; i < n;
       i += gridDim.x * blockDim.x) {
    float a = m1[i], b = m2[i];
    s += kld_e_f(a, sp_f(a), b, sp_f(b));
  }
  __shared__ float red[256];
  red[threadIdx.x] = s; __syncthreads();
  for (int off = 128; off > 0; off >>= 1) {
    if (threadIdx.x < off) red[threadIdx.x] += red[threadIdx.x + off];
    __syncthreads();
  }
  if (threadIdx.x == 0) atomicAdd(out, 0.5f * red[0]);
}

// per-slice symmetric KL: out[s] = 0.5*sum(kld(obs,enc)+kld(enc,obs)) over inner
__global__ void va_sym(const float* __restrict__ obs, const float* __restrict__ enc,
                       long long inner, float* __restrict__ out) {
  const float* e = enc + (long long)blockIdx.x * inner;
  float s = 0.f;
  for (long long i = threadIdx.x; i < inner; i += blockDim.x) {
    float om = obs[i], em = e[i];
    float os = sp_f(om), es = sp_f(em);
    s += kld_e_f(om, os, em, es) + kld_e_f(em, es, om, os);
  }
  __shared__ float red[256];
  red[threadIdx.x] = s; __syncthreads();
  for (int off = 128; off > 0; off >>= 1) {
    if (threadIdx.x < off) red[threadIdx.x] += red[threadIdx.x + off];
    __syncthreads();
  }
  if (threadIdx.x == 0) out[blockIdx.x] = 0.5f * red[0];
}

// argmin over small vector; optionally set or accumulate the min value
__global__ void va_argmin(const float* __restrict__ v, int n, int* __restrict__ jout,
                          float* set_min, float* add_min) {
  if (threadIdx.x == 0 && blockIdx.x == 0) {
    float mn = v[0]; int j = 0;
    for (int i = 1; i < n; ++i) { float x = v[i]; if (x < mn) { mn = x; j = i; } }
    *jout = j;
    if (set_min) *set_min = mn;
    if (add_min) *add_min += mn;
  }
}

// out += 0.5 * sum kld_e over dynamically-indexed slices (single block)
__global__ void va_kld_dyn(const float* __restrict__ m1b, const float* __restrict__ m2b,
                           const int* __restrict__ jptr, int baseIdx, long long stride,
                           int n, float* __restrict__ out) {
  long long s0 = (long long)(*jptr + baseIdx) * stride;
  const float* m1 = m1b + s0;
  const float* m2 = m2b + s0;
  float s = 0.f;
  for (int i = threadIdx.x; i < n; i += blockDim.x) {
    float a = m1[i], b = m2[i];
    s += kld_e_f(a, sp_f(a), b, sp_f(b));
  }
  __shared__ float red[256];
  red[threadIdx.x] = s; __syncthreads();
  for (int off = 128; off > 0; off >>= 1) {
    if (threadIdx.x < off) red[threadIdx.x] += red[threadIdx.x + off];
    __syncthreads();
  }
  if (threadIdx.x == 0) *out += 0.5f * red[0];
}

// dst[0:n] = src[(*jptr + baseIdx)*stride : +n]
__global__ void va_gather(const float* __restrict__ src, const int* __restrict__ jptr,
                          int baseIdx, long long stride, float* __restrict__ dst, int n) {
  long long s0 = (long long)(*jptr + baseIdx) * stride;
  int i = blockIdx.x * blockDim.x + threadIdx.x;
  if (i < n) dst[i] = src[s0 + i];
}

__global__ void va_zero5(float* p) { if (threadIdx.x < 5) p[threadIdx.x] = 0.f; }

// ---------------------------------------------------------------------------
extern "C" void kernel_launch(void* const* d_in, const int* in_sizes, int n_in,
                              void* d_out, int out_size, void* d_ws, size_t ws_size,
                              hipStream_t stream) {
  (void)in_sizes; (void)n_in; (void)out_size; (void)ws_size;
  const int T = 32, B = 512, F = 512, H = 512, Z = 256, G = 3, A = 10, NN = 3, C = 1000;

  // ---- inputs (setup_inputs order) ----
  const float* feat = (const float*)d_in[0];
  const float* h0   = (const float*)d_in[1];
  const float* eps_z    = (const float*)d_in[2];
  const float* eps_goal = (const float*)d_in[3];
  const float* eps_next = (const float*)d_in[4];
  const float* eps_fut  = (const float*)d_in[5];
  const float* Wpx = (const float*)d_in[6],  *bpx = (const float*)d_in[7];
  const float* Wenc1 = (const float*)d_in[8],  *benc1 = (const float*)d_in[9];
  const float* Wenc2 = (const float*)d_in[10], *benc2 = (const float*)d_in[11];
  const float* Wpr1 = (const float*)d_in[12], *bpr1 = (const float*)d_in[13];
  const float* Wpr2 = (const float*)d_in[14], *bpr2 = (const float*)d_in[15];
  const float* Wpz  = (const float*)d_in[16], *bpz  = (const float*)d_in[17];
  const float* Wih1 = (const float*)d_in[18], *Whh1 = (const float*)d_in[19];
  const float* bih1 = (const float*)d_in[20], *bhh1 = (const float*)d_in[21];
  const float* Wzn  = (const float*)d_in[22], *bzn  = (const float*)d_in[23];
  const float* Wprn1 = (const float*)d_in[24], *bprn1 = (const float*)d_in[25];
  const float* Wprn2 = (const float*)d_in[26], *bprn2 = (const float*)d_in[27];
  const float* Wga  = (const float*)d_in[28], *bga  = (const float*)d_in[29];
  const float* Woa1 = (const float*)d_in[30], *boa1 = (const float*)d_in[31];
  const float* Woa2 = (const float*)d_in[32], *boa2 = (const float*)d_in[33];
  const float* Wna1 = (const float*)d_in[34], *bna1 = (const float*)d_in[35];
  const float* Wna2 = (const float*)d_in[36], *bna2 = (const float*)d_in[37];
  const float* Wen  = (const float*)d_in[38], *ben  = (const float*)d_in[39];
  const float* Wact = (const float*)d_in[40], *bact = (const float*)d_in[41];
  const float* Wih2 = (const float*)d_in[42], *Whh2 = (const float*)d_in[43];
  const float* bih2 = (const float*)d_in[44], *bhh2 = (const float*)d_in[45];
  const float* Wcc  = (const float*)d_in[46], *bcc  = (const float*)d_in[47];
  const float* Wnc  = (const float*)d_in[48], *bnc  = (const float*)d_in[49];

  // ---- outputs ----
  float* dout       = (float*)d_out;
  float* out_cur    = dout;                       // [B,C]
  float* out_future = dout + (size_t)B * C;       // [NN,B,C]
  float* scal       = dout + (size_t)(NN + 1) * B * C;  // 5 scalars
  float* kld_lat_goal = scal + 0;
  float* kld_next_p   = scal + 1;
  float* lat_goal_dis = scal + 2;
  float* kld_future_p = scal + 3;
  float* future_dis_p = scal + 4;

  // ---- workspace bump allocator ----
  float* ws = (float*)d_ws;
  size_t off = 0;
  auto alloc = [&](size_t n) { float* p = ws + off; off += (n + 63) & ~(size_t)63; return p; };

  float* phix    = alloc((size_t)T * B * H);
  float* h       = alloc((size_t)B * H);
  float* cat2H   = alloc((size_t)B * 2 * H);
  float* tmpH    = alloc((size_t)B * H);
  float* enc     = alloc((size_t)B * Z);
  float* pri     = alloc((size_t)B * Z);
  float* zt      = alloc((size_t)B * Z);
  float* zproj   = alloc((size_t)B * H);
  float* gi      = alloc((size_t)B * 3 * H);
  float* gh      = alloc((size_t)B * 3 * H);
  float* obs_mean = alloc((size_t)B * Z);
  float* obs_goal = alloc((size_t)G * B * Z);
  float* pn_h     = alloc((size_t)B * Z);
  float* pn_h_bc  = alloc((size_t)G * B * Z);
  float* t1       = alloc((size_t)G * B * H);
  float* catGA    = alloc((size_t)G * B * (H + Z));
  float* obs_act  = alloc((size_t)G * B * H);
  float* next_mean = alloc((size_t)G * B * H);
  float* next_act  = alloc((size_t)G * A * B * H);   // later reused as fact
  float* bigScr    = alloc((size_t)G * A * B * H);   // mlp hidden / concat scratch
  float* pnext     = alloc((size_t)G * A * B * Z);   // later reused as pf
  float* obs_act_z = alloc((size_t)G * B * Z);
  float* oaz_bc    = alloc((size_t)G * A * B * Z);   // later reused as preZ_bc
  float* enc_next  = alloc((size_t)G * A * B * Z);   // later reused as encf
  float* naf    = alloc((size_t)B * H);
  float* prnZ   = alloc((size_t)B * Z);
  float* fmean  = alloc((size_t)B * H);
  float* preZ   = alloc((size_t)B * Z);
  float* fb0    = alloc((size_t)B * H);
  float* fb1    = alloc((size_t)B * H);
  float* pa     = alloc((size_t)B * H);
  float* pb     = alloc((size_t)B * H);
  float* symGA  = alloc(64);
  float* symF   = alloc(64);
  int*   jbuf   = (int*)alloc(64);

  // ---- launch helpers ----
  auto GEMM = [&](const float* X, const float* Wm, const float* bv, float* Y,
                  int M, int N, int K, bool relu) {
    dim3 g((unsigned)((N + BN - 1) / BN), (unsigned)(M / BM));
    if (relu) va_gemm<1><<<g, 256, 0, stream>>>(X, Wm, bv, Y, M, N, K);
    else      va_gemm<0><<<g, 256, 0, stream>>>(X, Wm, bv, Y, M, N, K);
  };
  auto EB = [](long long n) { return dim3((unsigned)((n + 255) / 256)); };

  // ---- init scalars, state ----
  va_zero5<<<1, 32, 0, stream>>>(scal);
  hipMemcpyAsync(h, h0, (size_t)B * H * sizeof(float), hipMemcpyDeviceToDevice, stream);

  // ---- phix = relu(feat @ Wpx + bpx) for all timesteps at once ----
  GEMM(feat, Wpx, bpx, phix, T * B, H, F, true);

  // ---- T-step GRU scan ----
  for (int t = 0; t < T; ++t) {
    const float* xp = phix + (size_t)t * B * H;
    va_concat2<<<EB((long long)B * 2 * H), 256, 0, stream>>>(h, H, xp, H, cat2H, (long long)B * 2 * H);
    GEMM(cat2H, Wenc1, benc1, tmpH, B, H, 2 * H, true);
    GEMM(tmpH, Wenc2, benc2, enc, B, Z, H, false);
    GEMM(h, Wpr1, bpr1, tmpH, B, H, H, true);
    GEMM(tmpH, Wpr2, bpr2, pri, B, Z, H, false);
    va_kld_sum<<<64, 256, 0, stream>>>(enc, pri, B * Z, kld_lat_goal);
    va_reparam<<<EB((long long)B * Z), 256, 0, stream>>>(enc, eps_z + (size_t)t * B * Z, zt, 1, (long long)B * Z, (long long)B * Z);
    GEMM(zt, Wpz, bpz, zproj, B, H, Z, true);
    va_concat2<<<EB((long long)B * 2 * H), 256, 0, stream>>>(xp, H, zproj, H, cat2H, (long long)B * 2 * H);
    GEMM(cat2H, Wih1, bih1, gi, B, 3 * H, 2 * H, false);
    GEMM(h, Whh1, bhh1, gh, B, 3 * H, H, false);
    va_gru<<<EB((long long)B * H), 256, 0, stream>>>(gi, gh, h, B, H);
  }

  // ---- goal stage ----
  GEMM(h, Wpr1, bpr1, tmpH, B, H, H, true);
  GEMM(tmpH, Wpr2, bpr2, obs_mean, B, Z, H, false);
  va_reparam<<<EB((long long)G * B * Z), 256, 0, stream>>>(obs_mean, eps_goal, obs_goal, G, (long long)B * Z, (long long)G * B * Z);
  GEMM(h, Wprn1, bprn1, tmpH, B, H, H, true);
  GEMM(tmpH, Wprn2, bprn2, pn_h, B, Z, H, false);
  va_bcast_mid<<<EB((long long)G * B * Z), 256, 0, stream>>>(pn_h, pn_h_bc, G, (long long)B * Z, (long long)G * B * Z);
  GEMM(obs_goal, Wzn, bzn, t1, G * B, H, Z, true);
  va_concat2<<<EB((long long)G * B * (H + Z)), 256, 0, stream>>>(t1, H, pn_h_bc, Z, catGA, (long long)G * B * (H + Z));
  GEMM(catGA, Wga, bga, obs_act, G * B, H, H + Z, true);
  va_concat2<<<EB((long long)G * B * (Z + H)), 256, 0, stream>>>(pn_h_bc, Z, obs_act, H, catGA, (long long)G * B * (Z + H));
  GEMM(catGA, Wga, bga, next_mean, G * B, H, Z + H, true);
  va_reparam<<<EB((long long)G * A * B * H), 256, 0, stream>>>(next_mean, eps_next, next_act, A, (long long)B * H, (long long)G * A * B * H);
  GEMM(next_act, Wna1, bna1, bigScr, G * A * B, H, H, true);
  GEMM(bigScr, Wna2, bna2, pnext, G * A * B, Z, H, false);
  GEMM(obs_act, Woa1, boa1, t1, G * B, H, H, true);
  GEMM(t1, Woa2, boa2, obs_act_z, G * B, Z, H, false);
  va_bcast_mid<<<EB((long long)G * A * B * Z), 256, 0, stream>>>(obs_act_z, oaz_bc, A, (long long)B * Z, (long long)G * A * B * Z);
  va_concat2<<<EB((long long)G * A * B * 2 * Z), 256, 0, stream>>>(pnext, Z, oaz_bc, Z, bigScr, (long long)G * A * B * 2 * Z);
  GEMM(bigScr, Wen, ben, enc_next, G * A * B, Z, 2 * Z, true);
  va_sym<<<G * A, 256, 0, stream>>>(obs_mean, enc_next, (long long)B * Z, symGA);
  va_argmin<<<1, 32, 0, stream>>>(symGA + (G - 1) * A, A, jbuf, lat_goal_dis, nullptr);
  va_kld_dyn<<<1, 256, 0, stream>>>(enc_next, pnext, jbuf, (G - 1) * A, (long long)B * Z, B * Z, kld_next_p);
  va_gather<<<EB((long long)B * H), 256, 0, stream>>>(next_act, jbuf, (G - 1) * A, (long long)B * H, naf, B * H);
  const float* obs_act_final = obs_act + (size_t)(G - 1) * B * H;

  // ---- future rollout (NN-1 sequential steps); ha lives in h ----
  const float* pre_ptr = obs_act_final;
  const float* cur_ptr = naf;
  float* fact = next_act;      // reuse (next_act dead after gather)
  float* pf   = pnext;         // reuse
  float* encf = enc_next;      // reuse
  float* preZ_bc = oaz_bc;     // reuse
  float* fcur[2] = { fb0, fb1 };
  for (int i = 0; i < NN - 1; ++i) {
    GEMM(pre_ptr, Wact, bact, pa, B, H, H, true);
    GEMM(cur_ptr, Wact, bact, pb, B, H, H, true);
    va_concat2<<<EB((long long)B * 2 * H), 256, 0, stream>>>(pa, H, pb, H, cat2H, (long long)B * 2 * H);
    GEMM(cat2H, Wih2, bih2, gi, B, 3 * H, 2 * H, false);
    GEMM(h, Whh2, bhh2, gh, B, 3 * H, H, false);
    va_gru<<<EB((long long)B * H), 256, 0, stream>>>(gi, gh, h, B, H);
    const float* pre_new = cur_ptr;   // pre = cur
    GEMM(h, Wprn1, bprn1, tmpH, B, H, H, true);
    GEMM(tmpH, Wprn2, bprn2, prnZ, B, Z, H, false);
    va_concat2<<<EB((long long)B * (Z + H)), 256, 0, stream>>>(prnZ, Z, pre_new, H, cat2H, (long long)B * (Z + H));
    GEMM(cat2H, Wga, bga, fmean, B, H, Z + H, true);
    va_reparam<<<EB((long long)A * B * H), 256, 0, stream>>>(fmean, eps_fut + (size_t)i * A * B * H, fact, A, (long long)B * H, (long long)A * B * H);
    GEMM(fact, Wna1, bna1, bigScr, A * B, H, H, true);
    GEMM(bigScr, Wna2, bna2, pf, A * B, Z, H, false);
    GEMM(pre_new, Wna1, bna1, tmpH, B, H, H, true);
    GEMM(tmpH, Wna2, bna2, preZ, B, Z, H, false);
    va_bcast_mid<<<EB((long long)A * B * Z), 256, 0, stream>>>(preZ, preZ_bc, A, (long long)B * Z, (long long)A * B * Z);
    va_concat2<<<EB((long long)A * B * 2 * Z), 256, 0, stream>>>(pf, Z, preZ_bc, Z, bigScr, (long long)A * B * 2 * Z);
    GEMM(bigScr, Wen, ben, encf, A * B, Z, 2 * Z, true);
    va_sym<<<A, 256, 0, stream>>>(obs_mean, encf, (long long)B * Z, symF);
    va_argmin<<<1, 32, 0, stream>>>(symF, A, jbuf, nullptr, future_dis_p);
    va_kld_dyn<<<1, 256, 0, stream>>>(encf, pf, jbuf, 0, (long long)B * Z, B * Z, kld_future_p);
    va_gather<<<EB((long long)B * H), 256, 0, stream>>>(fact, jbuf, 0, (long long)B * H, fcur[i], B * H);
    pre_ptr = pre_new;
    cur_ptr = fcur[i];
    GEMM(cur_ptr, Wnc, bnc, out_future + (size_t)(i + 1) * B * C, B, C, H, false);
  }

  // ---- heads ----
  GEMM(naf, Wnc, bnc, out_future, B, C, H, false);          // out_future[0] = out_next
  GEMM(obs_act_final, Wcc, bcc, out_cur, B, C, H, false);   // out_cur
}
